// GQAAttentionLayer_37177236914309
// MI455X (gfx1250) — compile-verified, matrix operand order checked by reading, not run on hardware
//
#include <hip/hip_runtime.h>

// ---------------------------------------------------------------------------
// GQA attention layer for MI455X (gfx1250, wave32, WMMA).
// All matmuls on v_wmma_f32_16x16x32_bf16 (f32 accumulate).
// Shared B / K / V tiles staged in LDS via async global->LDS DMA (ASYNCcnt),
// double-buffered so WMMAs overlap the next tile's fill. Fragment loads are
// batched ahead of the WMMA bursts to collapse ds waits.
// ---------------------------------------------------------------------------

#define D_MODEL 2048
#define TSEQ    2048
#define NHEADS  32
#define NKV     8
#define HDIM    64

typedef __attribute__((ext_vector_type(16))) __bf16 v16bf;
typedef __attribute__((ext_vector_type(8)))  float  v8f;
typedef __attribute__((ext_vector_type(4)))  int    v4i;

union FragU { v16bf v; uint4 u[2]; };

#if defined(__gfx1250__) && \
    __has_builtin(__builtin_amdgcn_global_load_async_to_lds_b128) && \
    __has_builtin(__builtin_amdgcn_s_wait_asynccnt)
#define USE_ASYNC 1
#else
#define USE_ASYNC 0
#endif

// 16-byte global -> LDS copy. Async DMA path when available (ASYNCcnt),
// else synchronous load+ds_store fallback.
__device__ __forceinline__ void copy16_g2l(const __bf16* gp, __bf16* lp) {
#if USE_ASYNC
  __builtin_amdgcn_global_load_async_to_lds_b128(
      (__attribute__((address_space(1))) v4i*)gp,
      (__attribute__((address_space(3))) v4i*)lp, 0, 0);
#else
  *(uint4*)lp = *(const uint4*)gp;
#endif
}

// Load one 16x32 bf16 WMMA fragment (A layout; B layout identical with
// "row" = output column). Per CDNA5 ISA 7.12.2:
//   lane l holds row (l%16); VGPR0-3 = K [kb, kb+8), VGPR4-7 = K [kb+16, kb+24)
//   where kb = (l<16) ? 0 : 8.  Two 16B vector loads per lane.
__device__ __forceinline__ v16bf load_frag(const __bf16* tile, int stride) {
  const int lane = threadIdx.x & 31;
  const int r    = lane & 15;
  const int kb   = (lane >> 4) << 3;   // 0 or 8
  const uint4* p = reinterpret_cast<const uint4*>(tile + (size_t)r * stride + kb);
  FragU f;
  f.u[0] = p[0];   // K: kb .. kb+7
  f.u[1] = p[2];   // K: kb+16 .. kb+23
  return f.v;
}

__device__ __forceinline__ v8f wmma_bf16(v16bf a, v16bf b, v8f c) {
  // (neg_a, A, neg_b, B, c_mod, C, reuse_a, reuse_b)
  return __builtin_amdgcn_wmma_f32_16x16x32_bf16(
      false, a, false, b, (short)0, c, false, false);
}

// ---------------------------------------------------------------------------
// Elementwise fp32 -> bf16 convert
__global__ __launch_bounds__(256) void cvt_bf16_kernel(
    const float* __restrict__ in, __bf16* __restrict__ out, int n) {
  int i = blockIdx.x * 256 + threadIdx.x;
  if (i < n) out[i] = (__bf16)in[i];
}

// Transpose + convert:  in [K][N] fp32  ->  out [N][K] bf16
__global__ __launch_bounds__(256) void transpose_bf16_kernel(
    const float* __restrict__ in, __bf16* __restrict__ out, int K, int N) {
  int n = blockIdx.x * 16 + (threadIdx.x & 15);
  int k = blockIdx.y * 16 + (threadIdx.x >> 4);
  if (n < N && k < K) out[(size_t)n * K + k] = (__bf16)in[(size_t)k * N + n];
}

// ---------------------------------------------------------------------------
// WMMA GEMM:  C[M][N] (f32) = A[M][K] (bf16) * BT[N][K] (bf16, i.e. B^T)
// Block = 128 threads = 4 waves; block tile = 128x64; wave tile = 32x64.
// B tile (64 cols x 32 K = 4KB) is shared by all 4 waves: staged in LDS via
// async DMA, double-buffered (next k-step fills while current one computes).
__global__ __launch_bounds__(128) void gemm_bf16_kernel(
    const __bf16* __restrict__ A, const __bf16* __restrict__ BT,
    float* __restrict__ C, int M, int N, int K) {
  const int tid  = threadIdx.x;
  const int w    = tid >> 5;
  const int lane = tid & 31;
  const int m0   = blockIdx.y * 128 + w * 32;
  const int n0   = blockIdx.x * 64;

  __shared__ __align__(16) __bf16 Blds[2][64][32];   // [buf][col][k] 8KB

  // Fill one B tile: 64 rows x 64B; 128 threads x 2 x 16B segments.
  auto fill_b = [&](int bufI, int k0) {
#pragma unroll
    for (int i = 0; i < 2; ++i) {
      const int s = tid + i * 128;         // 16B segment id, 0..255
      const int n = s >> 2;                // column (BT row), 0..63
      const int c = (s & 3) << 3;          // element offset in row
      copy16_g2l(BT + (size_t)(n0 + n) * K + k0 + c, &Blds[bufI][n][c]);
    }
  };

  v8f acc[2][4] = {};
  const __bf16* Arow0 = A + (size_t)m0 * K;
  const __bf16* Arow1 = A + (size_t)(m0 + 16) * K;

  fill_b(0, 0);                            // prologue: ASYNCcnt += 2 per wave

  int buf = 0;
  for (int k0 = 0; k0 < K; k0 += 32, buf ^= 1) {
    // Prefetch next B tile into the other buffer, then wait for current.
    if (k0 + 32 < K) {
      fill_b(buf ^ 1, k0 + 32);
#if USE_ASYNC
      __builtin_amdgcn_s_wait_asynccnt(2); // current tile's 2 DMAs retired
#endif
    } else {
#if USE_ASYNC
      __builtin_amdgcn_s_wait_asynccnt(0);
#endif
    }
    v16bf a0 = load_frag(Arow0 + k0, K);   // A frags stream from global/L2
    v16bf a1 = load_frag(Arow1 + k0, K);
    __syncthreads();                       // all waves' fills of `buf` landed

    // Batch all B-fragment LDS reads, then fire the 8 WMMAs back-to-back.
    v16bf bfr[4];
#pragma unroll
    for (int nb = 0; nb < 4; ++nb)
      bfr[nb] = load_frag(&Blds[buf][nb * 16][0], 32);
#pragma unroll
    for (int nb = 0; nb < 4; ++nb) {
      acc[0][nb] = wmma_bf16(a0, bfr[nb], acc[0][nb]);
      acc[1][nb] = wmma_bf16(a1, bfr[nb], acc[1][nb]);
    }
    __syncthreads();                       // reads of `buf` done -> refillable
  }

  const int r  = lane & 15;
  const int rb = (lane >> 4) << 3;
#pragma unroll
  for (int i = 0; i < 2; ++i)
#pragma unroll
    for (int nb = 0; nb < 4; ++nb)
#pragma unroll
      for (int j = 0; j < 8; ++j)
        C[(size_t)(m0 + i * 16 + rb + j) * N + (n0 + nb * 16 + r)] =
            acc[i][nb][j];
}

// ---------------------------------------------------------------------------
// RoPE + repack. One block per token t, 1024 threads.
//   q: fp32 [t][32*64] -> bf16 [h][t][64], rotated, pre-scaled by 1/sqrt(64)
//   k: fp32 [t][ 8*64] -> bf16 [h][t][64], rotated
//   v: fp32 [t][ 8*64] -> bf16 [h][64][t]  (transposed for PV B-fragments)
__global__ __launch_bounds__(1024) void rope_pack_kernel(
    const float* __restrict__ qf, const float* __restrict__ kf,
    const float* __restrict__ vf, const float* __restrict__ fc,
    __bf16* __restrict__ qbf, __bf16* __restrict__ kbf, __bf16* __restrict__ vT) {
  const int t   = blockIdx.x;
  const int tid = threadIdx.x;
  {
    const int hh = tid >> 5, i = tid & 31;
    const float c = fc[t * 64 + 2 * i], s = fc[t * 64 + 2 * i + 1];
    const float re = qf[(size_t)t * (NHEADS * HDIM) + hh * HDIM + 2 * i];
    const float im = qf[(size_t)t * (NHEADS * HDIM) + hh * HDIM + 2 * i + 1];
    const size_t base = ((size_t)hh * TSEQ + t) * HDIM + 2 * i;
    qbf[base]     = (__bf16)((re * c - im * s) * 0.125f);  // fold 1/sqrt(64)
    qbf[base + 1] = (__bf16)((re * s + im * c) * 0.125f);
  }
  if (tid < NKV * 32) {
    const int hh = tid >> 5, i = tid & 31;
    const float c = fc[t * 64 + 2 * i], s = fc[t * 64 + 2 * i + 1];
    const float re = kf[(size_t)t * (NKV * HDIM) + hh * HDIM + 2 * i];
    const float im = kf[(size_t)t * (NKV * HDIM) + hh * HDIM + 2 * i + 1];
    const size_t base = ((size_t)hh * TSEQ + t) * HDIM + 2 * i;
    kbf[base]     = (__bf16)(re * c - im * s);
    kbf[base + 1] = (__bf16)(re * s + im * c);
  }
  if (tid < NKV * HDIM) {
    const int hh = tid >> 6, d = tid & 63;
    vT[((size_t)hh * HDIM + d) * TSEQ + t] =
        (__bf16)vf[(size_t)t * (NKV * HDIM) + hh * HDIM + d];
  }
}

// ---------------------------------------------------------------------------
// Flash attention (causal, online softmax), WMMA throughout.
// Grid: (T/64 q-blocks, 32 heads). Block: 128 threads = 4 waves.
// K chunk (32x64) and V chunk (64x32) are shared by all 4 waves: staged in
// LDS via async DMA, double-buffered one chunk ahead.
__global__ __launch_bounds__(128) void attn_kernel(
    const __bf16* __restrict__ qbf, const __bf16* __restrict__ kbf,
    const __bf16* __restrict__ vT, __bf16* __restrict__ ybf) {
  const int qblk = blockIdx.x;
  const int h    = blockIdx.y;
  const int hkv  = h >> 2;                  // N_REP = 4
  const int tid  = threadIdx.x;
  const int w    = tid >> 5;
  const int lane = tid & 31;
  const int r    = lane & 15;
  const int rb   = (lane >> 4) << 3;

  __shared__ __align__(16) __bf16 Kc[2][32][HDIM];  // [buf][token][dim] 8KB
  __shared__ __align__(16) __bf16 Vc[2][HDIM][32];  // [buf][dim][token] 8KB
  __shared__ float Slds[4][16][32];
  __shared__ __align__(16) __bf16 Plds[4][16][32];
  __shared__ float mrow[64], lrow[64], crow[64];

  // Fill K+V chunk tiles (4 DMA issues per wave per fill).
  auto fill_kv = [&](int bufI, int kBase) {
#pragma unroll
    for (int i = 0; i < 2; ++i) {
      const int s = tid + i * 128;
      {  // K tile: 32 token-rows x 128B
        const int row = s >> 3, c = (s & 7) << 3;
        copy16_g2l(kbf + (((size_t)hkv * TSEQ) + kBase + row) * HDIM + c,
                   &Kc[bufI][row][c]);
      }
      {  // V tile: 64 dim-rows x 64B
        const int row = s >> 2, c = (s & 3) << 3;
        copy16_g2l(vT + ((size_t)hkv * HDIM + row) * TSEQ + kBase + c,
                   &Vc[bufI][row][c]);
      }
    }
  };

  fill_kv(0, 0);                            // prologue: ASYNCcnt += 4 per wave

  if (tid < 64) { mrow[tid] = -1e30f; lrow[tid] = 0.0f; }

  // Q fragments: 16 rows x 64 dims (two 16x32 K-fragments), loaded once.
  const __bf16* qtile =
      qbf + (((size_t)h * TSEQ) + (size_t)qblk * 64 + w * 16) * HDIM;
  v16bf qa0 = load_frag(qtile,      HDIM);
  v16bf qa1 = load_frag(qtile + 32, HDIM);

  v8f o[4] = {};  // O accumulator: 16 rows x 64 dims (f32)

  const int nChunks = 2 * (qblk + 1);
  int buf = 0;
  for (int it = 0; it < nChunks; ++it, buf ^= 1) {
    const int kBase = it * 32;

    // Prefetch next K/V chunk; wait for current one.
    if (it + 1 < nChunks) {
      fill_kv(buf ^ 1, kBase + 32);
#if USE_ASYNC
      __builtin_amdgcn_s_wait_asynccnt(4);
#endif
    } else {
#if USE_ASYNC
      __builtin_amdgcn_s_wait_asynccnt(0);
#endif
    }
    __syncthreads();   // (A) current K/V tiles visible to all waves

    // ---- S = Q * K^T (16 q-rows x 32 keys), 4 WMMAs, B-frags from LDS.
    // Batch all K-fragment reads first, then the WMMA burst.
    v16bf kb0[2], kb1[2];
#pragma unroll
    for (int nb = 0; nb < 2; ++nb) {
      kb0[nb] = load_frag(&Kc[buf][nb * 16][0],  HDIM);  // dims  0..31
      kb1[nb] = load_frag(&Kc[buf][nb * 16][32], HDIM);  // dims 32..63
    }
#pragma unroll
    for (int nb = 0; nb < 2; ++nb) {
      v8f s = {};
      s = wmma_bf16(qa0, kb0[nb], s);
      s = wmma_bf16(qa1, kb1[nb], s);
#pragma unroll
      for (int j = 0; j < 8; ++j) Slds[w][rb + j][nb * 16 + r] = s[j];
    }
    __syncthreads();   // (1) S tile in LDS

    // ---- Online softmax: thread t<64 owns one query row
    if (tid < 64) {
      const int row = tid, ww = row >> 4, rl = row & 15;
      const int tq = qblk * 64 + row;
      const float m_old = mrow[row];
      float mx = m_old;
      float sv[32];
#pragma unroll
      for (int c = 0; c < 32; ++c) {
        float s = Slds[ww][rl][c];
        sv[c] = (kBase + c > tq) ? -1e30f : s;   // causal mask
        mx = fmaxf(mx, sv[c]);
      }
      const float corr = __expf(m_old - mx);
      float lsum = lrow[row] * corr;
#pragma unroll
      for (int c = 0; c < 32; ++c) {
        const float pe = __expf(sv[c] - mx);
        lsum += pe;
        Plds[ww][rl][c] = (__bf16)pe;
      }
      mrow[row] = mx; lrow[row] = lsum; crow[row] = corr;
    }
    __syncthreads();   // (2) P, corr ready

    // ---- Rescale O by row correction factors
    float cr[8];
#pragma unroll
    for (int j = 0; j < 8; ++j) cr[j] = crow[w * 16 + rb + j];
#pragma unroll
    for (int j = 0; j < 8; ++j) {
      o[0][j] *= cr[j]; o[1][j] *= cr[j]; o[2][j] *= cr[j]; o[3][j] *= cr[j];
    }

    // ---- O += P * V : A = P (16x32 from LDS), B = V tile from LDS.
    // Batch P + all V fragments, then the WMMA burst.
    v16bf pa = load_frag(&Plds[w][0][0], 32);
    v16bf vfr[4];
#pragma unroll
    for (int nb = 0; nb < 4; ++nb)
      vfr[nb] = load_frag(&Vc[buf][nb * 16][0], 32);
#pragma unroll
    for (int nb = 0; nb < 4; ++nb)
      o[nb] = wmma_bf16(pa, vfr[nb], o[nb]);
    __syncthreads();   // (3) all reads of `buf` done -> refill safe next iter
  }

  // ---- Normalize by row sums, emit bf16 [t][32*64] for the output GEMM
  float li[8];
#pragma unroll
  for (int j = 0; j < 8; ++j) li[j] = 1.0f / lrow[w * 16 + rb + j];
#pragma unroll
  for (int nb = 0; nb < 4; ++nb)
#pragma unroll
    for (int j = 0; j < 8; ++j) {
      const size_t row_g = (size_t)qblk * 64 + w * 16 + rb + j;
      ybf[row_g * D_MODEL + h * HDIM + nb * 16 + r] =
          (__bf16)(o[nb][j] * li[j]);
    }
}

// ---------------------------------------------------------------------------
extern "C" void kernel_launch(void* const* d_in, const int* in_sizes, int n_in,
                              void* d_out, int out_size, void* d_ws, size_t ws_size,
                              hipStream_t stream) {
  (void)in_sizes; (void)n_in; (void)out_size; (void)ws_size;

  const float* x  = (const float*)d_in[0];
  const float* fc = (const float*)d_in[1];   // [T][32][2] cos/sin
  const float* wq = (const float*)d_in[2];
  const float* wk = (const float*)d_in[3];
  const float* wv = (const float*)d_in[4];
  const float* wo = (const float*)d_in[5];
  float* out = (float*)d_out;

  char* base = (char*)d_ws;
  size_t off = 0;
  auto alloc = [&](size_t bytes) -> void* {
    void* q = base + off;
    off += (bytes + 255) & ~(size_t)255;
    return q;
  };

  __bf16* xbf  = (__bf16*)alloc((size_t)TSEQ * D_MODEL * 2);        // 8 MB
  __bf16* wqT  = (__bf16*)alloc((size_t)D_MODEL * D_MODEL * 2);     // 8 MB
  __bf16* wkT  = (__bf16*)alloc((size_t)(NKV * HDIM) * D_MODEL * 2);
  __bf16* wvT  = (__bf16*)alloc((size_t)(NKV * HDIM) * D_MODEL * 2);
  __bf16* woT  = (__bf16*)alloc((size_t)D_MODEL * D_MODEL * 2);     // 8 MB
  float*  qf   = (float*)alloc((size_t)TSEQ * (NHEADS * HDIM) * 4); // 16 MB
  float*  kf   = (float*)alloc((size_t)TSEQ * (NKV * HDIM) * 4);
  float*  vf   = (float*)alloc((size_t)TSEQ * (NKV * HDIM) * 4);
  __bf16* qbf  = (__bf16*)alloc((size_t)NHEADS * TSEQ * HDIM * 2);  // 8 MB
  __bf16* kbf  = (__bf16*)alloc((size_t)NKV * TSEQ * HDIM * 2);
  __bf16* vTbf = (__bf16*)alloc((size_t)NKV * HDIM * TSEQ * 2);
  __bf16* ybf  = (__bf16*)alloc((size_t)TSEQ * D_MODEL * 2);        // 8 MB

  // 1) precision convert / transpose
  cvt_bf16_kernel<<<(TSEQ * D_MODEL + 255) / 256, 256, 0, stream>>>(
      x, xbf, TSEQ * D_MODEL);
  transpose_bf16_kernel<<<dim3(D_MODEL / 16, D_MODEL / 16), 256, 0, stream>>>(
      wq, wqT, D_MODEL, D_MODEL);
  transpose_bf16_kernel<<<dim3((NKV * HDIM) / 16, D_MODEL / 16), 256, 0, stream>>>(
      wk, wkT, D_MODEL, NKV * HDIM);
  transpose_bf16_kernel<<<dim3((NKV * HDIM) / 16, D_MODEL / 16), 256, 0, stream>>>(
      wv, wvT, D_MODEL, NKV * HDIM);
  transpose_bf16_kernel<<<dim3(D_MODEL / 16, D_MODEL / 16), 256, 0, stream>>>(
      wo, woT, D_MODEL, D_MODEL);

  // 2) QKV projections (WMMA, async-LDS staged B)
  gemm_bf16_kernel<<<dim3((NHEADS * HDIM) / 64, TSEQ / 128), 128, 0, stream>>>(
      xbf, wqT, qf, TSEQ, NHEADS * HDIM, D_MODEL);
  gemm_bf16_kernel<<<dim3((NKV * HDIM) / 64, TSEQ / 128), 128, 0, stream>>>(
      xbf, wkT, kf, TSEQ, NKV * HDIM, D_MODEL);
  gemm_bf16_kernel<<<dim3((NKV * HDIM) / 64, TSEQ / 128), 128, 0, stream>>>(
      xbf, wvT, vf, TSEQ, NKV * HDIM, D_MODEL);

  // 3) RoPE + head-major repack (+ V transpose)
  rope_pack_kernel<<<TSEQ, 1024, 0, stream>>>(qf, kf, vf, fc, qbf, kbf, vTbf);

  // 4) causal flash attention (WMMA, async-LDS staged K/V)
  attn_kernel<<<dim3(TSEQ / 64, NHEADS), 128, 0, stream>>>(qbf, kbf, vTbf, ybf);

  // 5) output projection (WMMA) -> fp32 result
  gemm_bf16_kernel<<<dim3(D_MODEL / 64, TSEQ / 128), 128, 0, stream>>>(
      ybf, woT, out, TSEQ, D_MODEL, D_MODEL);
}
